// LayerNormLSTM_2405181686417
// MI455X (gfx1250) — matrix operand order repaired and checked by reading, not compiled, not run on hardware
//
#include <hip/hip_runtime.h>

#define TT 512
#define BB 64
#define HH 512
#define GG 2048
#define LL 2
#define EPSF 1e-5f

typedef __attribute__((ext_vector_type(16))) __bf16 v16bf;
typedef __attribute__((ext_vector_type(8)))  float  v8f;

union BF16x16 { v16bf v; uint4 q[2]; __bf16 h[16]; };
union BF16x8  { uint4 q;  __bf16 h[8];  };

// ---- LDS layout (bytes). 286592 B < 320 KB/WGP (CDNA5) ----
#define SM_W    0        // 2 mats * 8 tiles * 16 ksteps * 1024B  = 262144 (bf16 B-fragments)
#define SM_GIH  262144   // 16 x 132 f32 raw ih gates             = 8448
#define SM_GHH  270592   // 16 x 132 f32 raw hh gates             = 8448
#define SM_C    279040   // 16 x 32  f32 cell state               = 2048
#define SM_T    281088   // 16 x 32  f32 tanh(cy)                 = 2048
#define SM_O    283136   // 16 x 32  f32 sigmoid(o)               = 2048
#define SM_REDL 285184   // 16 x 4 x 4 f32 local partials         = 1024
#define SM_TOT1 286208   // 16 x 4 f32                            = 256
#define SM_TOT2 286464   // 16 x 2 f32                            = 128
#define SMEM_BYTES 286592
#define GSTR 132

// ---- workspace layout (bytes) ----
#define OFF_XBF  ((size_t)0)           // T*B*H bf16 = 32 MB
#define OFF_H0B  ((size_t)33554432)    // T*B*H bf16 = 32 MB (layer-0 hidden, masked)
#define OFF_HBUF ((size_t)67108864)    // B*H bf16   = 64 KB (current h)
#define OFF_RED1 ((size_t)67174400)    // 4*16*16*4 f32 = 16 KB
#define OFF_RED2 ((size_t)67190784)    // 4*16*16*2 f32 = 8 KB
#define OFF_CTRL ((size_t)67198976)    // barrier counters, 512 B

__device__ __forceinline__ float sigm_(float x) { return 1.0f / (1.0f + expf(-x)); }

// Sense-reversing barrier across the 16 WGs of one row-group (via L2 atomics).
__device__ __forceinline__ void group_barrier(unsigned* cnt, unsigned* gen) {
  __builtin_amdgcn_fence(__ATOMIC_RELEASE, "agent");
  __syncthreads();
  if (threadIdx.x == 0) {
    unsigned g = __atomic_load_n(gen, __ATOMIC_RELAXED);
    if (__atomic_fetch_add(cnt, 1u, __ATOMIC_ACQ_REL) == 15u) {
      __atomic_store_n(cnt, 0u, __ATOMIC_RELAXED);
      __atomic_fetch_add(gen, 1u, __ATOMIC_RELEASE);
    } else {
      while (__atomic_load_n(gen, __ATOMIC_ACQUIRE) == g) { __builtin_amdgcn_s_sleep(2); }
    }
  }
  __syncthreads();
  __builtin_amdgcn_fence(__ATOMIC_ACQUIRE, "agent");
}

__global__ void cvt_x_bf16(const float* __restrict__ src, __bf16* __restrict__ dst) {
  size_t i = ((size_t)blockIdx.x * blockDim.x + threadIdx.x) * 8;
  if (i >= (size_t)TT * BB * HH) return;
  float4 a = *(const float4*)(src + i);
  float4 b = *(const float4*)(src + i + 4);
  BF16x8 u;
  u.h[0]=(__bf16)a.x; u.h[1]=(__bf16)a.y; u.h[2]=(__bf16)a.z; u.h[3]=(__bf16)a.w;
  u.h[4]=(__bf16)b.x; u.h[5]=(__bf16)b.y; u.h[6]=(__bf16)b.z; u.h[7]=(__bf16)b.w;
  *(uint4*)(dst + i) = u.q;
}

__global__ __launch_bounds__(256, 1) void lstm_persist(
    const float* __restrict__ Wih,  const float* __restrict__ Whh,
    const float* __restrict__ bih,  const float* __restrict__ bhh,
    const float* __restrict__ g_ih, const float* __restrict__ b_ih_ln,
    const float* __restrict__ g_hh, const float* __restrict__ b_hh_ln,
    const float* __restrict__ g_ho, const float* __restrict__ b_ho_ln,
    const int*   __restrict__ seq_lens,
    const __bf16* __restrict__ xbf,
    __bf16* __restrict__ h0b,
    __bf16* __restrict__ hbuf,
    float* __restrict__ red1, float* __restrict__ red2,
    unsigned* __restrict__ ctrl,
    float* __restrict__ out) {
  extern __shared__ char smem[];
  float* sgih  = (float*)(smem + SM_GIH);
  float* sghh  = (float*)(smem + SM_GHH);
  float* scb   = (float*)(smem + SM_C);
  float* stb   = (float*)(smem + SM_T);
  float* sob   = (float*)(smem + SM_O);
  float* sredl = (float*)(smem + SM_REDL);
  float* stot1 = (float*)(smem + SM_TOT1);
  float* stot2 = (float*)(smem + SM_TOT2);

  const int tid  = threadIdx.x;
  const int lane = tid & 31;
  const int wave = tid >> 5;             // 8 waves; wave == gate tile tt (0..7)
  const int rg   = blockIdx.x >> 4;      // row group (16 batch rows)
  const int cg   = blockIdx.x & 15;      // 32 hidden-column strip
  unsigned* bcnt = ctrl + rg * 32;
  unsigned* bgen = ctrl + rg * 32 + 16;

  const int mrow = lane & 15;
  const int half = lane >> 4;
  const int colbase = ((wave >> 1) << 9) + (cg << 5) + ((wave & 1) << 4);

  for (int layer = 0; layer < LL; ++layer) {
    const float* Wih_l = Wih + (size_t)layer * GG * HH;
    const float* Whh_l = Whh + (size_t)layer * GG * HH;

    // --- swizzle this WG's 128-gate-column weight strips into LDS B-fragment layout
    // B fragment (16x16x32 bf16): lane L -> N = L%16, elems 0..7 -> K = (L/16)*8 + e, 8..15 -> K = 16 + (L/16)*8 + (e-8)
    for (int idx = tid; idx < 8192; idx += 256) {
      int lane_ = idx & 31, ks = (idx >> 5) & 15, tt = (idx >> 9) & 7, mat = (idx >> 12) & 1;
      int N  = ((tt >> 1) << 9) + (cg << 5) + ((tt & 1) << 4) + (lane_ & 15);
      int k0 = (ks << 5) + ((lane_ >> 4) << 3);
      const float* src = (mat ? Whh_l : Wih_l) + (size_t)N * HH + k0;
      BF16x16 u;
#pragma unroll
      for (int i2 = 0; i2 < 16; ++i2) u.h[i2] = (__bf16)src[i2 + ((i2 >> 3) << 3)];
      uint4* dst = (uint4*)(smem + SM_W + ((((mat << 3) + tt) * 16 + ks) << 10) + (lane_ << 5));
      dst[0] = u.q[0]; dst[1] = u.q[1];
    }
    // --- zero cell state and our strip of the h broadcast buffer
    for (int e = tid; e < 512; e += 256) {
      int row = e >> 5, hc = e & 31;
      scb[row * 32 + hc] = 0.0f;
      hbuf[(size_t)(rg * 16 + row) * HH + (cg << 5) + hc] = (__bf16)0.0f;
    }
    __syncthreads();
    group_barrier(bcnt, bgen);

    // --- per-layer hoisted parameters
    int eRow[2], eHc[2], eRowg[2], eSlen[2];
    float pGih[2][4], pBihL[2][4], pGhh[2][4], pBhhL[2][4], pGho[2], pBho[2];
#pragma unroll
    for (int j = 0; j < 2; ++j) {
      int e = tid + (j << 8);
      eRow[j] = e >> 5; eHc[j] = e & 31; eRowg[j] = rg * 16 + eRow[j];
      eSlen[j] = seq_lens[eRowg[j]];
      int hcg = (cg << 5) + eHc[j];
      pGho[j] = g_ho[layer * HH + hcg]; pBho[j] = b_ho_ln[layer * HH + hcg];
#pragma unroll
      for (int q = 0; q < 4; ++q) {
        int gc = layer * GG + (q << 9) + hcg;
        pGih[j][q] = g_ih[gc]; pBihL[j][q] = b_ih_ln[gc];
        pGhh[j][q] = g_hh[gc]; pBhhL[j][q] = b_hh_ln[gc];
      }
    }
    const float bihL = bih[layer * GG + colbase + mrow];
    const float bhhL = bhh[layer * GG + colbase + mrow];
    const __bf16* xsrc = layer ? h0b : xbf;

    for (int t = 0; t < TT; ++t) {
      // ================= GEMM: ih and hh tiles for this wave =================
      v8f accI = {0.f,0.f,0.f,0.f,0.f,0.f,0.f,0.f};
      v8f accH = {0.f,0.f,0.f,0.f,0.f,0.f,0.f,0.f};
      const __bf16* ax = xsrc + (((size_t)(t << 6) + rg * 16 + mrow) << 9);
      const __bf16* ah = hbuf + ((size_t)(rg * 16 + mrow) << 9);
      const int aoff = half << 3;  // A layout: elems 0..7 -> K=half*8.., 8..15 -> K=16+half*8..
#pragma unroll 4
      for (int ks = 0; ks < 16; ++ks) {
        int k0 = (ks << 5) + aoff;
        BF16x16 fa, fh, bi, bh2;
        fa.q[0] = *(const uint4*)(ax + k0); fa.q[1] = *(const uint4*)(ax + k0 + 16);
        fh.q[0] = *(const uint4*)(ah + k0); fh.q[1] = *(const uint4*)(ah + k0 + 16);
        const uint4* wi = (const uint4*)(smem + SM_W + (((wave) * 16 + ks) << 10) + (lane << 5));
        const uint4* wh = (const uint4*)(smem + SM_W + (((8 + wave) * 16 + ks) << 10) + (lane << 5));
        bi.q[0]  = wi[0]; bi.q[1]  = wi[1];
        bh2.q[0] = wh[0]; bh2.q[1] = wh[1];
        accI = __builtin_amdgcn_wmma_f32_16x16x32_bf16(false, fa.v, false, bi.v,  (short)0, accI, false, false);
        accH = __builtin_amdgcn_wmma_f32_16x16x32_bf16(false, fh.v, false, bh2.v, (short)0, accH, false, false);
      }
      // dump raw gates (+bias, pre-LN) to LDS. D layout: lane -> N=lane%16, M = r + 8*(lane/16)
      {
        int m0 = half << 3, n = mrow;
#pragma unroll
        for (int r = 0; r < 8; ++r) {
          sgih[(m0 + r) * GSTR + (wave << 4) + n] = accI[r] + bihL;
          sghh[(m0 + r) * GSTR + (wave << 4) + n] = accH[r] + bhhL;
        }
      }
      __syncthreads();
      // ============== LN stats over the 2048 gate axis (distributed) ==============
      if (tid < 64) {
        int row = tid >> 2, seg = tid & 3;
        float sI = 0, qI = 0, sH = 0, qH = 0;
        const float* gi = sgih + row * GSTR + seg * 32;
        const float* gh = sghh + row * GSTR + seg * 32;
        for (int c2 = 0; c2 < 32; ++c2) {
          float a = gi[c2]; sI += a; qI += a * a;
          float b = gh[c2]; sH += b; qH += b * b;
        }
        float* d = sredl + (row * 4 + seg) * 4;
        d[0] = sI; d[1] = qI; d[2] = sH; d[3] = qH;
      }
      __syncthreads();
      if (tid < 16) {
        float s0 = 0, s1 = 0, s2 = 0, s3 = 0;
        for (int seg = 0; seg < 4; ++seg) {
          const float* d = sredl + (tid * 4 + seg) * 4;
          s0 += d[0]; s1 += d[1]; s2 += d[2]; s3 += d[3];
        }
        float* g = red1 + (((rg * 16 + cg) * 16) + tid) * 4;
        g[0] = s0; g[1] = s1; g[2] = s2; g[3] = s3;
      }
      group_barrier(bcnt, bgen);
      if (tid < 64) {
        int row = tid >> 2, st = tid & 3;
        float s = 0;
        for (int c2 = 0; c2 < 16; ++c2) s += red1[(((rg * 16 + c2) * 16) + row) * 4 + st];
        stot1[row * 4 + st] = s;
      }
      __syncthreads();
      // ============== gates -> cell ==============
#pragma unroll
      for (int j = 0; j < 2; ++j) {
        int row = eRow[j], hc = eHc[j];
        float mI = stot1[row * 4 + 0] * (1.0f / GG);
        float vI = stot1[row * 4 + 1] * (1.0f / GG) - mI * mI;
        float rI = rsqrtf(vI + EPSF);
        float mH = stot1[row * 4 + 2] * (1.0f / GG);
        float vH = stot1[row * 4 + 3] * (1.0f / GG) - mH * mH;
        float rH = rsqrtf(vH + EPSF);
        float gate[4];
#pragma unroll
        for (int q = 0; q < 4; ++q) {
          float a = sgih[row * GSTR + q * 32 + hc];
          float b = sghh[row * GSTR + q * 32 + hc];
          gate[q] = (a - mI) * rI * pGih[j][q] + pBihL[j][q] +
                    (b - mH) * rH * pGhh[j][q] + pBhhL[j][q];
        }
        float ii = sigm_(gate[0]), ff = sigm_(gate[1]);
        float gg = tanhf(gate[2]), oo = sigm_(gate[3]);
        float cold = scb[row * 32 + hc];
        float cy = ff * cold + ii * gg;
        float mt = (t < eSlen[j]) ? 1.0f : 0.0f;
        scb[row * 32 + hc] = cy * mt;
        stb[row * 32 + hc] = tanhf(cy);
        sob[row * 32 + hc] = oo;
      }
      __syncthreads();
      // ============== LN stats over H=512 of tanh(cy) (distributed) ==============
      if (tid < 16) {
        float s = 0, q = 0;
        const float* tb = stb + tid * 32;
        for (int c2 = 0; c2 < 32; ++c2) { float a = tb[c2]; s += a; q += a * a; }
        float* g = red2 + (((rg * 16 + cg) * 16) + tid) * 2;
        g[0] = s; g[1] = q;
      }
      group_barrier(bcnt, bgen);
      if (tid < 32) {
        int row = tid >> 1, st = tid & 1;
        float s = 0;
        for (int c2 = 0; c2 < 16; ++c2) s += red2[(((rg * 16 + c2) * 16) + row) * 2 + st];
        stot2[row * 2 + st] = s;
      }
      __syncthreads();
      // ============== output LN, h, writes ==============
#pragma unroll
      for (int j = 0; j < 2; ++j) {
        int row = eRow[j], hc = eHc[j], rowg = eRowg[j];
        float m = stot2[row * 2 + 0] * (1.0f / HH);
        float v = stot2[row * 2 + 1] * (1.0f / HH) - m * m;
        float r = rsqrtf(v + EPSF);
        float th = stb[row * 32 + hc];
        float hv = sob[row * 32 + hc] * ((th - m) * r * pGho[j] + pBho[j]);
        float mt = (t < eSlen[j]) ? 1.0f : 0.0f;
        float hy = hv * mt;
        int hcg = (cg << 5) + hc;
        hbuf[(size_t)rowg * HH + hcg] = (__bf16)hy;
        if (layer == 0) {
          h0b[(((size_t)(t << 6)) + rowg) * HH + hcg] = (__bf16)hy;
        } else {
          out[(((size_t)(t << 6)) + rowg) * HH + hcg] = hy;  // y
        }
        if (t == eSlen[j] - 1) {
          size_t yb = (size_t)TT * BB * HH;
          out[yb + ((size_t)(layer * BB + rowg)) * HH + hcg] = hy;                        // hy
          out[yb + (size_t)LL * BB * HH + ((size_t)(layer * BB + rowg)) * HH + hcg] =
              scb[row * 32 + hc];                                                         // cy
        }
      }
      group_barrier(bcnt, bgen);  // h_t visible before next step's GEMM
    }
  }
}

extern "C" void kernel_launch(void* const* d_in, const int* in_sizes, int n_in,
                              void* d_out, int out_size, void* d_ws, size_t ws_size,
                              hipStream_t stream) {
  const float* x        = (const float*)d_in[0];
  const int*   seq_lens = (const int*)d_in[1];
  const float* Wih      = (const float*)d_in[2];
  const float* Whh      = (const float*)d_in[3];
  const float* bih      = (const float*)d_in[4];
  const float* bhh      = (const float*)d_in[5];
  const float* g_ih     = (const float*)d_in[6];
  const float* b_ih_ln  = (const float*)d_in[7];
  const float* g_hh     = (const float*)d_in[8];
  const float* b_hh_ln  = (const float*)d_in[9];
  const float* g_ho     = (const float*)d_in[10];
  const float* b_ho_ln  = (const float*)d_in[11];

  char* ws = (char*)d_ws;
  __bf16*   xbf  = (__bf16*)(ws + OFF_XBF);
  __bf16*   h0b  = (__bf16*)(ws + OFF_H0B);
  __bf16*   hbuf = (__bf16*)(ws + OFF_HBUF);
  float*    red1 = (float*)(ws + OFF_RED1);
  float*    red2 = (float*)(ws + OFF_RED2);
  unsigned* ctrl = (unsigned*)(ws + OFF_CTRL);

  hipMemsetAsync(ws + OFF_CTRL, 0, 512, stream);
  cvt_x_bf16<<<8192, 256, 0, stream>>>(x, xbf);

  hipFuncSetAttribute(reinterpret_cast<const void*>(lstm_persist),
                      hipFuncAttributeMaxDynamicSharedMemorySize, SMEM_BYTES);
  lstm_persist<<<64, 256, SMEM_BYTES, stream>>>(
      Wih, Whh, bih, bhh, g_ih, b_ih_ln, g_hh, b_hh_ln, g_ho, b_ho_ln,
      seq_lens, xbf, h0b, hbuf, red1, red2, ctrl, (float*)d_out);
}
